// TemporalLSTM_67697274519654
// MI455X (gfx1250) — matrix operand order repaired and checked by reading, not compiled, not run on hardware
//
#include <hip/hip_runtime.h>

// ---------------- problem constants ----------------
constexpr int B_   = 1024;
constexpr int L_   = 96;
constexpr int D_   = 128;
constexpr int HS_  = 512;
constexpr int T_   = 24;
constexpr int N_   = 128;
constexpr int G4_  = 4 * HS_;          // 2048
constexpr float SCALE_ = 0.04419417382415922f; // 1/sqrt(512)

typedef __bf16 bf16;
typedef __attribute__((ext_vector_type(16))) __bf16 v16bf;
typedef __attribute__((ext_vector_type(8)))  float  v8f;

// ---------------- workspace layout (bytes) ----------------
constexpr size_t algn(size_t x) { return (x + 255) & ~(size_t)255; }
constexpr size_t OFF_HBF   = 0;                                              // (B*L, D) A-swizzled bf16
constexpr size_t OFF_WAB   = algn(OFF_HBF   + (size_t)B_*L_*D_*2);           // (T, D, N)  B-swizzled bf16
constexpr size_t OFF_UAB   = algn(OFF_WAB   + (size_t)T_*D_*N_*2);           // (T, HS, N) B-swizzled bf16
constexpr size_t OFF_WB    = algn(OFF_UAB   + (size_t)T_*HS_*N_*2);          // (T, D, 4HS) B-swizzled bf16
constexpr size_t OFF_UB    = algn(OFF_WB    + (size_t)T_*D_*G4_*2);          // (T, HS, 4HS) B-swizzled bf16
constexpr size_t OFF_CU    = algn(OFF_UB    + (size_t)T_*HS_*G4_*2);         // (B,N) f32
constexpr size_t OFF_SCORE = algn(OFF_CU    + (size_t)B_*N_*4);              // (B,L) f32
constexpr size_t OFF_BETA  = algn(OFF_SCORE + (size_t)B_*L_*4);              // (B,L) f32
constexpr size_t OFF_HTB   = algn(OFF_BETA  + (size_t)B_*L_*4);              // (B,D) A-swizzled bf16
constexpr size_t OFF_HF    = algn(OFF_HTB   + (size_t)B_*D_*2);              // (B,HS) f32
constexpr size_t OFF_CF    = algn(OFF_HF    + (size_t)B_*HS_*4);             // (B,HS) f32
constexpr size_t OFF_HB    = algn(OFF_CF    + (size_t)B_*HS_*4);             // (B,HS) A-swizzled bf16
constexpr size_t OFF_CB    = algn(OFF_HB    + (size_t)B_*HS_*2);             // (B,HS) A-swizzled bf16
constexpr size_t OFF_GATES = algn(OFF_CB    + (size_t)B_*HS_*2);             // (B,4HS) f32
constexpr size_t OFF_YPREV = algn(OFF_GATES + (size_t)B_*G4_*4);             // (B,) f32

// ---------------- fragment swizzle (writer & reader share these) ----------------
// A fragment (16x32, M x K): lane = (K-half-sel ? 16:0) + m; 16 bf16 contiguous per lane.
//   lanes 0-15 hold K 0-7 & 16-23 ; lanes 16-31 hold K 8-15 & 24-31 (ISA layout).
__device__ __forceinline__ size_t a_swz(int m, int k, int K) {
  int mt = m >> 4, kt = k >> 5;
  int ml = m & 15, kl = k & 31;
  int half = (kl >> 3) & 1;                      // 8-15 / 24-31 -> upper lanes
  int j    = (kl & 7) + ((kl & 16) >> 1);        // element index within 16
  int ktiles = K >> 5;
  return ((((size_t)mt * ktiles + kt) * 32) + half * 16 + ml) * 16 + j;
}
// B fragment (32x16, K x N): lanes 0-15 hold K 0-7 & 16-23 (N=lane), lanes 16-31 hold K 8-15 & 24-31.
__device__ __forceinline__ size_t b_swz(int k, int n, int N) {
  int kt = k >> 5, nt = n >> 4;
  int kl = k & 31, nl = n & 15;
  int half = (kl >> 3) & 1;
  int j    = (kl & 7) + ((kl & 16) >> 1);
  int ntiles = N >> 4;
  return ((((size_t)kt * ntiles + nt) * 32) + half * 16 + nl) * 16 + j;
}

// Per-lane contiguous 32B fragment loads -> 2x global_load_b128 each.
__device__ __forceinline__ v16bf load_a_sw(const bf16* __restrict__ src,
                                           int mt, int kt, int ktiles) {
  int lane = threadIdx.x & 31;
  return *(const v16bf*)(src + (((size_t)mt * ktiles + kt) * 32 + lane) * 16);
}
__device__ __forceinline__ v16bf load_b_sw(const bf16* __restrict__ src,
                                           int kt, int nt, int ntiles) {
  int lane = threadIdx.x & 31;
  return *(const v16bf*)(src + (((size_t)kt * ntiles + nt) * 32 + lane) * 16);
}

__device__ __forceinline__ v8f wmma_bf16(v16bf a, v16bf b, v8f c) {
  return __builtin_amdgcn_wmma_f32_16x16x32_bf16(false, a, false, b,
                                                 (short)0, c, false, false);
}

__device__ __forceinline__ float fast_tanh(float x) {
  float e = __expf(2.0f * x);
  return (e - 1.0f) / (e + 1.0f);
}
__device__ __forceinline__ float fast_sig(float x) {
  return 1.0f / (1.0f + __expf(-x));
}

// ---------------- conversion / init kernels ----------------
// H (M=B*L rows, K=D cols) f32 -> A-swizzled bf16
__global__ void cvt_a_swz(const float* __restrict__ in, bf16* __restrict__ out,
                          int K) {
  size_t i = (size_t)blockIdx.x * blockDim.x + threadIdx.x;
  int k = (int)(i % K);
  int m = (int)(i / K);
  out[a_swz(m, k, K)] = (bf16)in[i];
}

// weights (K, T, Nc) f32 -> per-t B-swizzled bf16 slices (stride K*Nc)
__global__ void cvt_b_swz(const float* __restrict__ in, bf16* __restrict__ out,
                          int K, int Nc) {
  size_t i = (size_t)blockIdx.x * blockDim.x + threadIdx.x;
  int n = (int)(i % Nc);
  size_t r = i / Nc;
  int t = (int)(r % T_);
  int k = (int)(r / T_);
  out[(size_t)t * K * Nc + b_swz(k, n, Nc)] = (bf16)in[i];
}

__global__ void init_state(const float* __restrict__ emb, const float* __restrict__ y0,
                           float* __restrict__ hf, float* __restrict__ cf,
                           bf16* __restrict__ hb, bf16* __restrict__ cb,
                           float* __restrict__ yprev) {
  int i = blockIdx.x * blockDim.x + threadIdx.x;   // B*HS threads
  int b = i >> 9, hs = i & (HS_ - 1);
  float h = emb[(size_t)b * 2 * HS_ + hs];
  float c = emb[(size_t)b * 2 * HS_ + HS_ + hs];
  hf[i] = h; cf[i] = c;
  size_t sw = a_swz(b, hs, HS_);
  hb[sw] = (bf16)h; cb[sw] = (bf16)c;
  if (hs == 0) yprev[b] = y0[b];
}

// ---------------- per-step kernels ----------------
// CU = c @ Ua_t : (1024 x 512) x (512 x 128); one wave per 16x32 output tile.
__global__ void cu_gemm(const bf16* __restrict__ Cst, const bf16* __restrict__ Uab,
                        float* __restrict__ CU) {
  int wave = (int)((blockIdx.x * blockDim.x + threadIdx.x) >> 5);
  int tm = wave >> 2;          // 64 M-tiles
  int tg = wave & 3;           // 4 groups of 2 N-tiles
  v8f acc[2] = {};
#pragma unroll
  for (int kt = 0; kt < HS_ / 32; ++kt) {
    v16bf a = load_a_sw(Cst, tm, kt, HS_ / 32);
#pragma unroll
    for (int q = 0; q < 2; ++q)
      acc[q] = wmma_bf16(a, load_b_sw(Uab, kt, tg * 2 + q, N_ / 16), acc[q]);
  }
  int lane = threadIdx.x & 31, half = lane >> 4, cc = lane & 15;
#pragma unroll
  for (int q = 0; q < 2; ++q)
#pragma unroll
    for (int r = 0; r < 8; ++r)
      CU[(size_t)(tm * 16 + r + half * 8) * N_ + (tg * 2 + q) * 16 + cc] = acc[q][r];
}

// score[b,l] = sum_n tanh( (H@wa)[b,l,n] + CU[b,n] + ba[n] ) * va[n] * scale
// one wave per 16-row tile of flattened (b,l); all 8 N-tiles held in registers.
__global__ void attn_score(const bf16* __restrict__ Hbf, const bf16* __restrict__ Wab,
                           const float* __restrict__ CU, const float* __restrict__ ba_t,
                           const float* __restrict__ Va, int t,
                           float* __restrict__ score) {
  int wave = (int)((blockIdx.x * blockDim.x + threadIdx.x) >> 5);  // 0..6143
  int b       = wave / 6;
  int rowBase = wave * 16;                 // = b*96 + l0
  int lane = threadIdx.x & 31, half = lane >> 4, cc = lane & 15;

  v8f acc[8] = {};
#pragma unroll
  for (int kt = 0; kt < D_ / 32; ++kt) {
    v16bf a = load_a_sw(Hbf, wave, kt, D_ / 32);
#pragma unroll
    for (int nt = 0; nt < 8; ++nt)
      acc[nt] = wmma_bf16(a, load_b_sw(Wab, kt, nt, N_ / 16), acc[nt]);
  }

  float sacc[8];
#pragma unroll
  for (int r = 0; r < 8; ++r) sacc[r] = 0.0f;
#pragma unroll
  for (int nt = 0; nt < 8; ++nt) {
    int n = nt * 16 + cc;
    float add = CU[(size_t)b * N_ + n] + ba_t[n];
    float vav = Va[(size_t)n * T_ + t];
#pragma unroll
    for (int r = 0; r < 8; ++r) sacc[r] += fast_tanh(acc[nt][r] + add) * vav;
  }
  // reduce over the 16 N-lanes inside each half (wave32 shuffles)
#pragma unroll
  for (int off = 1; off < 16; off <<= 1)
#pragma unroll
    for (int r = 0; r < 8; ++r) sacc[r] += __shfl_xor(sacc[r], off, 32);
  if (cc == 0)
#pragma unroll
    for (int r = 0; r < 8; ++r)
      score[rowBase + r + half * 8] = sacc[r] * SCALE_;
}

__global__ void softmax96(const float* __restrict__ score, float* __restrict__ beta) {
  __shared__ float red[128];
  int b = blockIdx.x, l = threadIdx.x;
  float v = (l < L_) ? score[(size_t)b * L_ + l] : -1e30f;
  red[l] = v; __syncthreads();
  for (int s = 64; s > 0; s >>= 1) { if (l < s) red[l] = fmaxf(red[l], red[l + s]); __syncthreads(); }
  float m = red[0]; __syncthreads();
  float e = (l < L_) ? __expf(v - m) : 0.0f;
  red[l] = e; __syncthreads();
  for (int s = 64; s > 0; s >>= 1) { if (l < s) red[l] += red[l + s]; __syncthreads(); }
  if (l < L_) beta[(size_t)b * L_ + l] = e / red[0];
}

__global__ void context(const float* __restrict__ H, const float* __restrict__ beta,
                        bf16* __restrict__ htb) {
  int b = blockIdx.x, d = threadIdx.x;  // 128 threads
  float s = 0.0f;
  for (int l = 0; l < L_; ++l)
    s += beta[(size_t)b * L_ + l] * H[((size_t)b * L_ + l) * D_ + d];
  htb[a_swz(b, d, D_)] = (bf16)s;       // write directly in A-fragment layout
}

// gates = ht@W_t + h@U_t + bias_t + yprev*wy_t : M=1024, N=2048, K=128+512
// one wave computes a 16x64 output strip (A fragment reused across 4 WMMAs).
__global__ void gates_gemm(const bf16* __restrict__ htb, const bf16* __restrict__ hb,
                           const bf16* __restrict__ Wb, const bf16* __restrict__ Ub,
                           const float* __restrict__ bias_t, const float* __restrict__ wy_t,
                           const float* __restrict__ yprev, float* __restrict__ gates) {
  int wave = (int)((blockIdx.x * blockDim.x + threadIdx.x) >> 5);
  int tm = wave >> 5;          // 64 M-tiles
  int tg = wave & 31;          // 32 groups of 4 N-tiles
  int tn = tg * 4;
  __builtin_prefetch(Ub + (size_t)tn * 32 * 16, 0, 1);   // -> global_prefetch_b8
  v8f acc[4] = {};
#pragma unroll
  for (int kt = 0; kt < D_ / 32; ++kt) {
    v16bf a = load_a_sw(htb, tm, kt, D_ / 32);
#pragma unroll
    for (int q = 0; q < 4; ++q)
      acc[q] = wmma_bf16(a, load_b_sw(Wb, kt, tn + q, G4_ / 16), acc[q]);
  }
#pragma unroll
  for (int kt = 0; kt < HS_ / 32; ++kt) {
    v16bf a = load_a_sw(hb, tm, kt, HS_ / 32);
#pragma unroll
    for (int q = 0; q < 4; ++q)
      acc[q] = wmma_bf16(a, load_b_sw(Ub, kt, tn + q, G4_ / 16), acc[q]);
  }
  int lane = threadIdx.x & 31, half = lane >> 4, cc = lane & 15;
#pragma unroll
  for (int q = 0; q < 4; ++q) {
    int col = (tn + q) * 16 + cc;
    float bj = bias_t[col], wyj = wy_t[col];
#pragma unroll
    for (int r = 0; r < 8; ++r) {
      int row = tm * 16 + r + half * 8;
      gates[(size_t)row * G4_ + col] = acc[q][r] + bj + yprev[row] * wyj;
    }
  }
}

__global__ void lstm_pointwise(const float* __restrict__ gates,
                               float* __restrict__ cf, bf16* __restrict__ cb,
                               float* __restrict__ hf, bf16* __restrict__ hb,
                               float* __restrict__ hidden_out, int t) {
  int i = blockIdx.x * blockDim.x + threadIdx.x;    // B*HS threads
  int b = i >> 9, hs = i & (HS_ - 1);
  const float* g = gates + (size_t)b * G4_;
  float ig = fast_sig(g[hs]);
  float fg = fast_sig(g[HS_ + hs]);
  float gg = fast_tanh(g[2 * HS_ + hs]);
  float og = fast_sig(g[3 * HS_ + hs]);
  float cn = fg * cf[i] + ig * gg;
  float hn = og * fast_tanh(cn);
  cf[i] = cn; hf[i] = hn;
  size_t sw = a_swz(b, hs, HS_);
  cb[sw] = (bf16)cn;                     // state kept in A-fragment layout
  hb[sw] = (bf16)hn;
  hidden_out[((size_t)b * T_ + t) * HS_ + hs] = hn;   // (B,T,HS)
}

__global__ void y_project(const float* __restrict__ hf, const float* __restrict__ fcw,
                          const float* __restrict__ fcb, int t,
                          float* __restrict__ yprev, float* __restrict__ yout) {
  __shared__ float red[256];
  int b = blockIdx.x, tid = threadIdx.x;
  float s = 0.0f;
  for (int hs = tid; hs < HS_; hs += 256)
    s += hf[(size_t)b * HS_ + hs] * fcw[(size_t)t * HS_ + hs];
  red[tid] = s; __syncthreads();
  for (int st = 128; st > 0; st >>= 1) { if (tid < st) red[tid] += red[tid + st]; __syncthreads(); }
  if (tid == 0) {
    float y = red[0] + fcb[t];
    yprev[b] = y;
    yout[(size_t)b * T_ + t] = y;                     // y_out is (B,T)
  }
}

// ---------------- host launcher ----------------
extern "C" void kernel_launch(void* const* d_in, const int* in_sizes, int n_in,
                              void* d_out, int out_size, void* d_ws, size_t ws_size,
                              hipStream_t stream) {
  const float* H    = (const float*)d_in[0];
  const float* y0   = (const float*)d_in[1];
  const float* emb  = (const float*)d_in[2];
  const float* Wa   = (const float*)d_in[4];
  const float* Ua   = (const float*)d_in[5];
  const float* ba   = (const float*)d_in[6];
  const float* Va   = (const float*)d_in[7];
  const float* W    = (const float*)d_in[8];
  const float* U    = (const float*)d_in[9];
  const float* bias = (const float*)d_in[10];
  const float* Wy   = (const float*)d_in[11];
  const float* fcw  = (const float*)d_in[12];
  const float* fcb  = (const float*)d_in[13];

  float* out = (float*)d_out;
  char*  ws  = (char*)d_ws;

  bf16*  Hbf   = (bf16*)(ws + OFF_HBF);
  bf16*  Wab   = (bf16*)(ws + OFF_WAB);
  bf16*  Uab   = (bf16*)(ws + OFF_UAB);
  bf16*  Wb    = (bf16*)(ws + OFF_WB);
  bf16*  Ub    = (bf16*)(ws + OFF_UB);
  float* CU    = (float*)(ws + OFF_CU);
  float* score = (float*)(ws + OFF_SCORE);
  float* beta  = (float*)(ws + OFF_BETA);
  bf16*  htb   = (bf16*)(ws + OFF_HTB);
  float* hf    = (float*)(ws + OFF_HF);
  float* cf    = (float*)(ws + OFF_CF);
  bf16*  hb    = (bf16*)(ws + OFF_HB);
  bf16*  cb    = (bf16*)(ws + OFF_CB);
  float* gates = (float*)(ws + OFF_GATES);
  float* yprev = (float*)(ws + OFF_YPREV);

  // one-time conversions into WMMA fragment layouts (re-done every launch; deterministic)
  cvt_a_swz<<<(B_ * L_ * D_) / 256, 256, 0, stream>>>(H, Hbf, D_);
  cvt_b_swz<<<((size_t)D_  * T_ * N_ ) / 256, 256, 0, stream>>>(Wa, Wab, D_,  N_);
  cvt_b_swz<<<((size_t)HS_ * T_ * N_ ) / 256, 256, 0, stream>>>(Ua, Uab, HS_, N_);
  cvt_b_swz<<<((size_t)D_  * T_ * G4_) / 256, 256, 0, stream>>>(W,  Wb,  D_,  G4_);
  cvt_b_swz<<<((size_t)HS_ * T_ * G4_) / 256, 256, 0, stream>>>(U,  Ub,  HS_, G4_);
  init_state<<<(B_ * HS_) / 256, 256, 0, stream>>>(emb, y0, hf, cf, hb, cb, yprev);

  for (int t = 0; t < T_; ++t) {
    cu_gemm   <<<32,  256, 0, stream>>>(cb, Uab + (size_t)t * HS_ * N_, CU);      // 256 waves
    attn_score<<<768, 256, 0, stream>>>(Hbf, Wab + (size_t)t * D_ * N_, CU,
                                        ba + (size_t)t * N_, Va, t, score);       // 6144 waves
    softmax96 <<<B_, 128, 0, stream>>>(score, beta);
    context   <<<B_, 128, 0, stream>>>(H, beta, htb);
    gates_gemm<<<256, 256, 0, stream>>>(htb, hb,
                                        Wb + (size_t)t * D_  * G4_,
                                        Ub + (size_t)t * HS_ * G4_,
                                        bias + (size_t)t * G4_,
                                        Wy   + (size_t)t * G4_,
                                        yprev, gates);                            // 2048 waves
    lstm_pointwise<<<(B_ * HS_) / 256, 256, 0, stream>>>(gates, cf, cb, hf, hb,
                                                         out + (size_t)B_ * T_, t);
    y_project <<<B_, 256, 0, stream>>>(hf, fcw, fcb, t, yprev, out);
  }
}
